// TopKRouter_65687229825575
// MI455X (gfx1250) — matrix operand order repaired and checked by reading, not compiled, not run on hardware
//
#include <hip/hip_runtime.h>
#include <hip/hip_bf16.h>

// -------- CDNA5 WMMA vector types --------
typedef __attribute__((ext_vector_type(16))) __bf16          v16bf;
typedef __attribute__((ext_vector_type(16))) unsigned short  v16u;
typedef __attribute__((ext_vector_type(8)))  unsigned int    v8u;
typedef __attribute__((ext_vector_type(8)))  float           v8f;

#define DMODEL   2048
#define NEXPERT  64
#define TOPK     2
#define WAVES_PER_BLOCK 4
#define ROWS_PER_WAVE   32                    // two 16-row A tiles share B
#define ROWS_PER_BLOCK  (WAVES_PER_BLOCK * ROWS_PER_WAVE)   // 128
#define KCHUNK   128                          // K-values staged per chunk
#define NCHUNK   (DMODEL / KCHUNK)            // 16
#define LDS_STRIDE 65                         // odd -> conflict-free softmax rows

// ---------------------------------------------------------------------------
// Pre-pass: split W (f32) into bf16 hi/lo planes in workspace.
// ---------------------------------------------------------------------------
__global__ void wsplit_kernel(const float* __restrict__ W,
                              unsigned short* __restrict__ Whi,
                              unsigned short* __restrict__ Wlo, int n) {
  int i = blockIdx.x * blockDim.x + threadIdx.x;
  if (i < n) {
    float f = W[i];
    unsigned int b = __float_as_uint(f);
    unsigned short h = (unsigned short)(b >> 16);
    float hf = __uint_as_float(((unsigned int)h) << 16);
    float l = f - hf;
    Whi[i] = h;
    Wlo[i] = (unsigned short)(__float_as_uint(l) >> 16);
  }
}

// ---------------------------------------------------------------------------
// Router kernel: one wave32 owns 32 rows of x and all 64 experts.
// W (bf16 hi/lo) staged block-wide through LDS via CDNA5 async copies
// (global_load_async_to_lds_b128 + s_wait_asynccnt), double buffered.
// logits = hi*hi + hi*lo + lo*hi via v_wmma_f32_16x16x32_bf16.
// ---------------------------------------------------------------------------
__global__ __launch_bounds__(WAVES_PER_BLOCK * 32)
void router_kernel(const float* __restrict__ x,
                   const unsigned short* __restrict__ Whi,
                   const unsigned short* __restrict__ Wlo,
                   float* __restrict__ probs,
                   int*   __restrict__ idx_out,
                   float* __restrict__ wt_out) {
  // Epilogue LDS aliases the W staging buffer (W no longer needed then).
  __shared__ union {
    unsigned short wbuf[2][2][NEXPERT][KCHUNK];  // 64 KB
    struct {
      float eLds[WAVES_PER_BLOCK][ROWS_PER_WAVE][LDS_STRIDE];
      float invS[WAVES_PER_BLOCK][ROWS_PER_WAVE];
    } ep;
  } smem;

  const int tid    = threadIdx.x;
  const int lane   = tid & 31;
  const int wv     = tid >> 5;
  const int m      = lane & 15;   // A row / B expert column within a tile
  const int hiHalf = lane >> 4;   // which K-group this lane carries
  const int rowBase = blockIdx.x * ROWS_PER_BLOCK + wv * ROWS_PER_WAVE;

  const float* xrow0 = x + (size_t)(rowBase + m) * DMODEL;
  const float* xrow1 = xrow0 + (size_t)16 * DMODEL;
  const int kA = hiHalf ? 16 : 0; // A: lanes16-31 carry K 16-31
  const int kB = hiHalf ? 16 : 0; // B: lanes16-31 carry K 16-31

  v8f acc[2][4] = {};             // [row-tile][expert-tile], 32 x 64 logits

  // ---- async stage of one 32 KB W chunk: 2048 x 16B segs, 16 per thread ----
  auto issue_chunk = [&](int buf, int kc) {
#pragma unroll
    for (int j = 0; j < 16; ++j) {
      const int seg   = tid + j * 128;
      const int plane = seg >> 10;          // 0 = hi, 1 = lo
      const int rem   = seg & 1023;
      const int row   = rem >> 4;           // expert row
      const int part  = rem & 15;           // 16B segment within 256B row
      const unsigned short* g =
          (plane ? Wlo : Whi) + (size_t)row * DMODEL + kc + part * 8;
      const unsigned l =
          (unsigned)(uintptr_t)&smem.wbuf[buf][plane][row][part * 8];
      asm volatile("global_load_async_to_lds_b128 %0, %1, off"
                   :: "v"(l), "v"(g) : "memory");
    }
  };

  // f32 row fragment -> error-free bf16 hi/lo split, packed u32 via v_perm_b32
  auto buildA = [&](const float* xr, v16bf& a_hi, v16bf& a_lo) {
    const uint4* pa = (const uint4*)xr;
    uint4 q0 = pa[0], q1 = pa[1];
    const uint4* pb = (const uint4*)(xr + 8);
    uint4 q2 = pb[0], q3 = pb[1];
    unsigned uf[16] = {q0.x, q0.y, q0.z, q0.w, q1.x, q1.y, q1.z, q1.w,
                       q2.x, q2.y, q2.z, q2.w, q3.x, q3.y, q3.z, q3.w};
    v8u hi, lo;
#pragma unroll
    for (int p = 0; p < 8; ++p) {
      const unsigned e0 = uf[2 * p], e1 = uf[2 * p + 1];
      // {bf16(f1), bf16(f0)} by truncation: pick high halves of both words
      hi[p] = __builtin_amdgcn_perm(e1, e0, 0x07060302u);
      const float l0 = __uint_as_float(e0) - __uint_as_float(e0 & 0xffff0000u);
      const float l1 = __uint_as_float(e1) - __uint_as_float(e1 & 0xffff0000u);
      lo[p] = __builtin_amdgcn_perm(__float_as_uint(l1), __float_as_uint(l0),
                                    0x07060302u);
    }
    a_hi = __builtin_bit_cast(v16bf, hi);
    a_lo = __builtin_bit_cast(v16bf, lo);
  };

  issue_chunk(0, 0);

  for (int c = 0; c < NCHUNK; ++c) {
    const int cur = c & 1;
    if (c + 1 < NCHUNK) {
      issue_chunk(cur ^ 1, (c + 1) * KCHUNK);            // prefetch next chunk
      asm volatile("s_wait_asynccnt 0x10" ::: "memory"); // current 16 done
    } else {
      asm volatile("s_wait_asynccnt 0x0" ::: "memory");
    }
    __syncthreads();   // whole block sees wbuf[cur]

#pragma unroll 1       // keep live ranges < 256 VGPRs (no s_set_vgpr_msb)
    for (int ks = 0; ks < KCHUNK / 32; ++ks) {
      const int k0 = c * KCHUNK + ks * 32;
      __builtin_prefetch((const void*)(xrow0 + k0 + 512), 0, 3);

      v16bf a_hi0, a_lo0, a_hi1, a_lo1;
      buildA(xrow0 + k0 + kA, a_hi0, a_lo0);
      buildA(xrow1 + k0 + kA, a_hi1, a_lo1);

#pragma unroll
      for (int t = 0; t < 4; ++t) {
        const int koff = ks * 32 + kB;
        v16bf b_hi = __builtin_bit_cast(
            v16bf, *(const v16u*)&smem.wbuf[cur][0][t * 16 + m][koff]);
        v16bf b_lo = __builtin_bit_cast(
            v16bf, *(const v16u*)&smem.wbuf[cur][1][t * 16 + m][koff]);
        acc[0][t] = __builtin_amdgcn_wmma_f32_16x16x32_bf16(
            false, a_hi0, false, b_hi, (short)0, acc[0][t], false, false);
        acc[0][t] = __builtin_amdgcn_wmma_f32_16x16x32_bf16(
            false, a_hi0, false, b_lo, (short)0, acc[0][t], false, false);
        acc[0][t] = __builtin_amdgcn_wmma_f32_16x16x32_bf16(
            false, a_lo0, false, b_hi, (short)0, acc[0][t], false, false);
        acc[1][t] = __builtin_amdgcn_wmma_f32_16x16x32_bf16(
            false, a_hi1, false, b_hi, (short)0, acc[1][t], false, false);
        acc[1][t] = __builtin_amdgcn_wmma_f32_16x16x32_bf16(
            false, a_hi1, false, b_lo, (short)0, acc[1][t], false, false);
        acc[1][t] = __builtin_amdgcn_wmma_f32_16x16x32_bf16(
            false, a_lo1, false, b_hi, (short)0, acc[1][t], false, false);
      }
    }
    __syncthreads();   // done reading wbuf[cur] before it is overwritten
  }

  // ---- scatter logits to LDS (C layout: vgpr j -> row j(+8), lane -> col) ----
#pragma unroll
  for (int u = 0; u < 2; ++u)
#pragma unroll
    for (int t = 0; t < 4; ++t)
#pragma unroll
      for (int j = 0; j < 8; ++j)
        smem.ep.eLds[wv][u * 16 + hiHalf * 8 + j][t * 16 + m] = acc[u][t][j];

  __syncthreads();

  // ---- per-row softmax + top-2: every lane owns one of the 32 rows ----
  {
    const int r = lane;
    float mx = -3.4e38f;
    for (int e = 0; e < NEXPERT; ++e) mx = fmaxf(mx, smem.ep.eLds[wv][r][e]);

    float s = 0.f;
    float tv0 = -3.4e38f, tv1 = -3.4e38f;
    int   ti0 = 0, ti1 = 0;
    for (int e = 0; e < NEXPERT; ++e) {
      float v = __expf(smem.ep.eLds[wv][r][e] - mx);
      smem.ep.eLds[wv][r][e] = v;   // stash exp for the prob write pass
      s += v;
      if (v > tv0)      { tv1 = tv0; ti1 = ti0; tv0 = v; ti0 = e; }
      else if (v > tv1) { tv1 = v;  ti1 = e; }
    }
    const float inv = 1.0f / s;
    smem.ep.invS[wv][r] = inv;

    const float p0 = tv0 * inv, p1 = tv1 * inv;
    const float wsum = p0 + p1 + 1e-9f;
    const int row = rowBase + r;
    idx_out[row * TOPK + 0] = ti0;
    idx_out[row * TOPK + 1] = ti1;
    wt_out[row * TOPK + 0] = p0 / wsum;
    wt_out[row * TOPK + 1] = p1 / wsum;
  }

  __syncthreads();

  // ---- coalesced router_probs writes: 32 rows x 64 experts per wave ----
  float* pout = probs + (size_t)rowBase * NEXPERT;
#pragma unroll
  for (int it = 0; it < 2 * ROWS_PER_WAVE; ++it) {
    const int r = it >> 1;
    const int e = ((it & 1) << 5) + lane;
    pout[r * NEXPERT + e] = smem.ep.eLds[wv][r][e] * smem.ep.invS[wv][r];
  }
}

// ---------------------------------------------------------------------------
extern "C" void kernel_launch(void* const* d_in, const int* in_sizes, int n_in,
                              void* d_out, int out_size, void* d_ws, size_t ws_size,
                              hipStream_t stream) {
  const float* x = (const float*)d_in[0];
  const float* W = (const float*)d_in[1];
  const int N  = in_sizes[0] / DMODEL;       // 16384
  const int WE = NEXPERT * DMODEL;           // 131072 elems

  unsigned short* Whi = (unsigned short*)d_ws;       // 256 KB
  unsigned short* Wlo = Whi + WE;                    // 256 KB

  // d_out layout: probs (N*64 f32) | indices (N*2 i32) | weights (N*2 f32)
  float* base    = (float*)d_out;
  float* probs   = base;
  int*   idx_out = (int*)(base + (size_t)N * NEXPERT);
  float* wt_out  = base + (size_t)N * NEXPERT + (size_t)N * TOPK;

  wsplit_kernel<<<(WE + 255) / 256, 256, 0, stream>>>(W, Whi, Wlo, WE);
  router_kernel<<<N / ROWS_PER_BLOCK, WAVES_PER_BLOCK * 32, 0, stream>>>(
      x, Whi, Wlo, probs, idx_out, wt_out);
}
